// GPTMLPMoE_86088324481592
// MI455X (gfx1250) — compile-verified, matrix operand order checked by reading, not run on hardware
//
#include <hip/hip_runtime.h>
#include <math.h>

// ---------------- problem constants (from reference setup_inputs) -----------
#define BATCH 4
#define SEQ   2048
#define NTOK  (BATCH*SEQ)     // 8192 tokens
#define DMODEL 1024
#define HFF    4096
#define NEXP   8
#define TOPK   2
#define CAP    2048           // ceil(1.0 * 2 * 8192 / 8)

typedef __attribute__((ext_vector_type(16))) __bf16 v16bf;
typedef __attribute__((ext_vector_type(8)))  float  v8f;
typedef int v4i __attribute__((vector_size(16)));

#if defined(__gfx1250__) && __has_builtin(__builtin_amdgcn_global_load_async_to_lds_b128) && __has_builtin(__builtin_amdgcn_s_wait_asynccnt)
#define USE_ASYNC 1
#else
#define USE_ASYNC 0
#endif

#define AS1V(p) ((__attribute__((address_space(1))) v4i*)(p))
#define AS3V(p) ((__attribute__((address_space(3))) v4i*)(p))

// f32 -> bf16, round-half-up (cheap: one add + shift)
__device__ __forceinline__ unsigned short f2bf(float f) {
    return (unsigned short)((__float_as_uint(f) + 0x8000u) >> 16);
}
__device__ __forceinline__ unsigned int pack2bf(float lo, float hi) {
    unsigned int a = __float_as_uint(lo) + 0x8000u;
    unsigned int b = __float_as_uint(hi) + 0x8000u;
    return (a >> 16) | (b & 0xffff0000u);
}

// ---------------------------------------------------------------------------
// Prep 1: elementwise f32 -> bf16 for x.  4 floats / thread.
// ---------------------------------------------------------------------------
__global__ void cvt_x_kernel(const float* __restrict__ x,
                             unsigned short* __restrict__ xbf) {
    size_t i = ((size_t)blockIdx.x * 256 + threadIdx.x) * 4;
    if (i >= (size_t)NTOK * DMODEL) return;
    float4 v = *(const float4*)(x + i);
    uint2 o;
    o.x = pack2bf(v.x, v.y);
    o.y = pack2bf(v.z, v.w);
    *(uint2*)(xbf + i) = o;
}

// ---------------------------------------------------------------------------
// Prep 2: per-expert transpose + convert:  in[e][R][C] f32 -> out[e][C][R] bf16
// 32x32 tile through LDS. grid: (C/32, R/32, E), 256 threads.
// ---------------------------------------------------------------------------
__global__ void tconv_kernel(const float* __restrict__ in,
                             unsigned short* __restrict__ out,
                             int R, int C) {
    __shared__ unsigned short t[32][36];
    const int e  = blockIdx.z;
    const int rb = blockIdx.y * 32;
    const int cb = blockIdx.x * 32;
    const int tid = threadIdx.x;
    const float* ip = in + (size_t)e * R * C;
    unsigned short* op = out + (size_t)e * R * C;

    {   // read 32x32 f32 tile (coalesced), convert, store to LDS
        int r  = tid >> 3;
        int c4 = (tid & 7) * 4;
        float4 v = *(const float4*)(ip + (size_t)(rb + r) * C + cb + c4);
        t[r][c4 + 0] = f2bf(v.x);
        t[r][c4 + 1] = f2bf(v.y);
        t[r][c4 + 2] = f2bf(v.z);
        t[r][c4 + 3] = f2bf(v.w);
    }
    __syncthreads();
    {   // write transposed bf16 (coalesced 8B stores)
        int c  = tid >> 3;
        int r4 = (tid & 7) * 4;
        uint2 o;
        o.x = (unsigned int)t[r4 + 0][c] | ((unsigned int)t[r4 + 1][c] << 16);
        o.y = (unsigned int)t[r4 + 2][c] | ((unsigned int)t[r4 + 3][c] << 16);
        *(uint2*)(op + (size_t)(cb + c) * R + rb + r4) = o;
    }
}

// ---------------------------------------------------------------------------
// Kernel: gating.  One wave per token: logits = x[n,:] @ wg, softmax over 8
// experts, top-2, normalized gate values.
// ---------------------------------------------------------------------------
__global__ void gate_kernel(const float* __restrict__ x,
                            const float* __restrict__ wg,
                            int* __restrict__ gidx,
                            float* __restrict__ gval) {
    const int lane = threadIdx.x & 31;
    const int wave = threadIdx.x >> 5;
    const int n = blockIdx.x * 8 + wave;
    if (n >= NTOK) return;

    float s[NEXP];
#pragma unroll
    for (int e = 0; e < NEXP; ++e) s[e] = 0.f;
    const float* xr = x + (size_t)n * DMODEL;
    for (int d = lane; d < DMODEL; d += 32) {
        float xv = xr[d];
        const float* wr = wg + (size_t)d * NEXP;
#pragma unroll
        for (int e = 0; e < NEXP; ++e) s[e] += xv * wr[e];
    }
#pragma unroll
    for (int off = 16; off > 0; off >>= 1) {
#pragma unroll
        for (int e = 0; e < NEXP; ++e) s[e] += __shfl_xor(s[e], off, 32);
    }
    if (lane == 0) {
        float m = s[0];
#pragma unroll
        for (int e = 1; e < NEXP; ++e) m = fmaxf(m, s[e]);
        float sum = 0.f, p[NEXP];
#pragma unroll
        for (int e = 0; e < NEXP; ++e) { p[e] = __expf(s[e] - m); sum += p[e]; }
        float inv = 1.f / sum;
#pragma unroll
        for (int e = 0; e < NEXP; ++e) p[e] *= inv;
        int i0 = 0; float v0 = p[0];
#pragma unroll
        for (int e = 1; e < NEXP; ++e) if (p[e] > v0) { v0 = p[e]; i0 = e; }
        int i1 = (i0 == 0) ? 1 : 0; float v1 = p[i1];
#pragma unroll
        for (int e = 0; e < NEXP; ++e)
            if (e != i0 && p[e] > v1) { v1 = p[e]; i1 = e; }
        float denom = fmaxf(v0 + v1, 1e-9f);
        gidx[n * 2 + 0] = i0;  gval[n * 2 + 0] = v0 / denom;
        gidx[n * 2 + 1] = i1;  gval[n * 2 + 1] = v1 / denom;
    }
}

// ---------------------------------------------------------------------------
// Kernel: GShard capacity scan (single workgroup, chunked prefix sums).
// ---------------------------------------------------------------------------
__global__ void scan_kernel(const int* __restrict__ gidx,
                            float* __restrict__ gval,
                            int* __restrict__ pos,
                            int* __restrict__ slot_token) {
    __shared__ int cnt[256 * NEXP];
    __shared__ int tot[NEXP];
    __shared__ int off[NEXP];
    const int tid = threadIdx.x;
    const int CH = NTOK / 256;

    for (int i = tid; i < NEXP * CAP; i += 256) slot_token[i] = -1;
    if (tid < NEXP) off[tid] = 0;
    __syncthreads();

    for (int k = 0; k < TOPK; ++k) {
        int lc[NEXP];
#pragma unroll
        for (int e = 0; e < NEXP; ++e) lc[e] = 0;
        for (int j = 0; j < CH; ++j) lc[gidx[(tid * CH + j) * 2 + k]]++;
#pragma unroll
        for (int e = 0; e < NEXP; ++e) cnt[tid * NEXP + e] = lc[e];
        __syncthreads();
        if (tid < NEXP) {
            int run = off[tid];
            for (int t = 0; t < 256; ++t) {
                int tmp = cnt[t * NEXP + tid];
                cnt[t * NEXP + tid] = run;
                run += tmp;
            }
            tot[tid] = run;
        }
        __syncthreads();
        int rp[NEXP];
#pragma unroll
        for (int e = 0; e < NEXP; ++e) rp[e] = cnt[tid * NEXP + e];
        for (int j = 0; j < CH; ++j) {
            int n = tid * CH + j;
            int e = gidx[n * 2 + k];
            int p = rp[e]++;
            bool keep = (p < CAP);
            pos[n * 2 + k] = keep ? p : 0;
            if (keep) slot_token[e * CAP + p] = n;
            else      gval[n * 2 + k] = 0.f;
        }
        __syncthreads();
        if (tid < NEXP) off[tid] = tot[tid];
        __syncthreads();
    }
}

// ---------------------------------------------------------------------------
// WMMA tiling: 256x128 tile / 256-thread block (8 waves), each wave computes
// 64x64 via 4x4 register blocking -> 16 WMMA per K-step vs 16 ds_load_b128
// (1:1 ratio).  bf16 16x16x32, f32 accumulate.  LDS pitch 40 ushorts (80B).
// __launch_bounds__(256, 1) releases the full VGPR budget (no acc spills).
// ---------------------------------------------------------------------------
#define LPITCH 40
#define KT 32
#define MT 256
#define NT 128

__device__ __forceinline__ v16bf frag_from_lds(const unsigned short* base,
                                               int row, int khalf) {
    union { unsigned int u[8]; v16bf v; } f;
    const unsigned short* r = base + row * LPITCH + khalf;
#pragma unroll
    for (int v = 0; v < 8; ++v) {
        int k = ((v & 4) ? 16 : 0) + (v & 3) * 2;
        f.u[v] = *(const unsigned int*)(r + k);
    }
    return f.v;
}

// stage one K-step: full A row per thread (64B), half B row per thread (32B)
__device__ __forceinline__ void stage_tiles(const unsigned short* ag,
                                            const unsigned short* bg,
                                            unsigned short* As,
                                            unsigned short* Bs,
                                            int tid) {
    unsigned short* ad = As + tid * LPITCH;
    unsigned short* bd = Bs + (tid >> 1) * LPITCH + (tid & 1) * 16;
#if USE_ASYNC
    __builtin_amdgcn_global_load_async_to_lds_b128(AS1V(ag),      AS3V(ad),      0, 0);
    __builtin_amdgcn_global_load_async_to_lds_b128(AS1V(ag + 8),  AS3V(ad + 8),  0, 0);
    __builtin_amdgcn_global_load_async_to_lds_b128(AS1V(ag + 16), AS3V(ad + 16), 0, 0);
    __builtin_amdgcn_global_load_async_to_lds_b128(AS1V(ag + 24), AS3V(ad + 24), 0, 0);
    __builtin_amdgcn_global_load_async_to_lds_b128(AS1V(bg),      AS3V(bd),      0, 0);
    __builtin_amdgcn_global_load_async_to_lds_b128(AS1V(bg + 8),  AS3V(bd + 8),  0, 0);
#else
    const uint4* as = (const uint4*)ag;
    const uint4* bs = (const uint4*)bg;
    ((uint4*)ad)[0] = as[0]; ((uint4*)ad)[1] = as[1];
    ((uint4*)ad)[2] = as[2]; ((uint4*)ad)[3] = as[3];
    ((uint4*)bd)[0] = bs[0]; ((uint4*)bd)[1] = bs[1];
#endif
}

// Kernel: fused dispatch-gather + GEMM1 + bias + exact GELU -> h (bf16)
// grid: (HFF/NT, CAP/MT, NEXP)
__global__ void __launch_bounds__(256, 1)
gemm1_kernel(const unsigned short* __restrict__ xbf,
             const unsigned short* __restrict__ w1t,
             const float* __restrict__ b1,
             const int* __restrict__ slot_token,
             unsigned short* __restrict__ hbuf) {
    __shared__ unsigned short As[2][MT * LPITCH];
    __shared__ unsigned short Bs[2][NT * LPITCH];

    const int e  = blockIdx.z;
    const int cb = blockIdx.y * MT;
    const int hb = blockIdx.x * NT;
    const int tid  = threadIdx.x;
    const int lane = tid & 31;
    const int wave = tid >> 5;
    const int wm = wave & 3;          // 4 bands of 64 rows
    const int wn = wave >> 2;         // 2 groups of 64 cols
    const int lr = lane & 15;
    const int kh = (lane >> 4) * 8;

    // loaders: A row = tid (gathered token), B row = tid>>1 (weight column)
    int tok = slot_token[e * CAP + cb + tid];
    if (tok < 0) tok = 0;  // empty slot: contained garbage, never combined
    const unsigned short* ag = xbf + (size_t)tok * DMODEL;
    const unsigned short* bg =
        w1t + ((size_t)e * HFF + hb + (tid >> 1)) * DMODEL + (tid & 1) * 16;

    v8f acc[4][4] = {};
    const int NIT = DMODEL / KT;

    stage_tiles(ag, bg, As[0], Bs[0], tid);
    for (int it = 0; it < NIT; ++it) {
        const int cur = it & 1;
#if USE_ASYNC
        if (it + 1 < NIT) {
            stage_tiles(ag + (it + 1) * KT, bg + (it + 1) * KT,
                        As[cur ^ 1], Bs[cur ^ 1], tid);
            __builtin_amdgcn_s_wait_asynccnt(6);
        } else {
            __builtin_amdgcn_s_wait_asynccnt(0);
        }
#else
        if (it + 1 < NIT)
            stage_tiles(ag + (it + 1) * KT, bg + (it + 1) * KT,
                        As[cur ^ 1], Bs[cur ^ 1], tid);
#endif
        __syncthreads();
        v16bf bfr[4];
#pragma unroll
        for (int sn = 0; sn < 4; ++sn)
            bfr[sn] = frag_from_lds(Bs[cur], wn * 64 + sn * 16 + lr, kh);
#pragma unroll
        for (int sm = 0; sm < 4; ++sm) {
            v16bf afr = frag_from_lds(As[cur], wm * 64 + sm * 16 + lr, kh);
#pragma unroll
            for (int sn = 0; sn < 4; ++sn)
                acc[sm][sn] = __builtin_amdgcn_wmma_f32_16x16x32_bf16(
                    false, afr, false, bfr[sn], (short)0, acc[sm][sn],
                    false, false);
        }
        __syncthreads();
    }

    const int rhi = (lane >> 4) * 8;
#pragma unroll
    for (int sm = 0; sm < 4; ++sm) {
#pragma unroll
        for (int sn = 0; sn < 4; ++sn) {
            int col = hb + wn * 64 + sn * 16 + lr;
            float bias = b1[e * HFF + col];
#pragma unroll
            for (int v = 0; v < 8; ++v) {
                int r = cb + wm * 64 + sm * 16 + rhi + v;
                float val = acc[sm][sn][v] + bias;
                val = 0.5f * val * (1.0f + erff(val * 0.70710678118654752f));
                hbuf[((size_t)e * CAP + r) * HFF + col] = f2bf(val);
            }
        }
    }
}

// Kernel: GEMM2 (h bf16 @ w2t bf16) + b2 -> y (f32)
// grid: (DMODEL/NT, CAP/MT, NEXP)
__global__ void __launch_bounds__(256, 1)
gemm2_kernel(const unsigned short* __restrict__ hbuf,
             const unsigned short* __restrict__ w2t,
             const float* __restrict__ b2,
             float* __restrict__ y) {
    __shared__ unsigned short As[2][MT * LPITCH];
    __shared__ unsigned short Bs[2][NT * LPITCH];

    const int e  = blockIdx.z;
    const int cb = blockIdx.y * MT;
    const int db = blockIdx.x * NT;
    const int tid  = threadIdx.x;
    const int lane = tid & 31;
    const int wave = tid >> 5;
    const int wm = wave & 3;
    const int wn = wave >> 2;
    const int lr = lane & 15;
    const int kh = (lane >> 4) * 8;

    const unsigned short* ag = hbuf + ((size_t)e * CAP + cb + tid) * HFF;
    const unsigned short* bg =
        w2t + ((size_t)e * DMODEL + db + (tid >> 1)) * HFF + (tid & 1) * 16;

    v8f acc[4][4] = {};
    const int NIT = HFF / KT;

    stage_tiles(ag, bg, As[0], Bs[0], tid);
    for (int it = 0; it < NIT; ++it) {
        const int cur = it & 1;
#if USE_ASYNC
        if (it + 1 < NIT) {
            stage_tiles(ag + (it + 1) * KT, bg + (it + 1) * KT,
                        As[cur ^ 1], Bs[cur ^ 1], tid);
            __builtin_amdgcn_s_wait_asynccnt(6);
        } else {
            __builtin_amdgcn_s_wait_asynccnt(0);
        }
#else
        if (it + 1 < NIT)
            stage_tiles(ag + (it + 1) * KT, bg + (it + 1) * KT,
                        As[cur ^ 1], Bs[cur ^ 1], tid);
#endif
        __syncthreads();
        v16bf bfr[4];
#pragma unroll
        for (int sn = 0; sn < 4; ++sn)
            bfr[sn] = frag_from_lds(Bs[cur], wn * 64 + sn * 16 + lr, kh);
#pragma unroll
        for (int sm = 0; sm < 4; ++sm) {
            v16bf afr = frag_from_lds(As[cur], wm * 64 + sm * 16 + lr, kh);
#pragma unroll
            for (int sn = 0; sn < 4; ++sn)
                acc[sm][sn] = __builtin_amdgcn_wmma_f32_16x16x32_bf16(
                    false, afr, false, bfr[sn], (short)0, acc[sm][sn],
                    false, false);
        }
        __syncthreads();
    }

    const int rhi = (lane >> 4) * 8;
#pragma unroll
    for (int sm = 0; sm < 4; ++sm) {
#pragma unroll
        for (int sn = 0; sn < 4; ++sn) {
            int col = db + wn * 64 + sn * 16 + lr;
            float bias = b2[e * DMODEL + col];
#pragma unroll
            for (int v = 0; v < 8; ++v) {
                int r = cb + wm * 64 + sm * 16 + rhi + v;
                y[((size_t)e * CAP + r) * DMODEL + col] = acc[sm][sn][v] + bias;
            }
        }
    }
}

// ---------------------------------------------------------------------------
// Kernel: combine. out[n,:] = g0*y[e0,p0,:] + g1*y[e1,p1,:]
// ---------------------------------------------------------------------------
__global__ void combine_kernel(const float* __restrict__ y,
                               const int* __restrict__ gidx,
                               const float* __restrict__ gval,
                               const int* __restrict__ pos,
                               float* __restrict__ out) {
    const int n = blockIdx.x;
    const int d = threadIdx.x * 4;
    const int e0 = gidx[n * 2 + 0], e1 = gidx[n * 2 + 1];
    const int p0 = pos[n * 2 + 0],  p1 = pos[n * 2 + 1];
    const float g0 = gval[n * 2 + 0], g1 = gval[n * 2 + 1];
    const float4 a = *(const float4*)&y[((size_t)e0 * CAP + p0) * DMODEL + d];
    const float4 b = *(const float4*)&y[((size_t)e1 * CAP + p1) * DMODEL + d];
    float4 o;
    o.x = g0 * a.x + g1 * b.x;
    o.y = g0 * a.y + g1 * b.y;
    o.z = g0 * a.z + g1 * b.z;
    o.w = g0 * a.w + g1 * b.w;
    *(float4*)&out[(size_t)n * DMODEL + d] = o;
}

// ---------------------------------------------------------------------------
extern "C" void kernel_launch(void* const* d_in, const int* in_sizes, int n_in,
                              void* d_out, int out_size, void* d_ws, size_t ws_size,
                              hipStream_t stream) {
    const float* x  = (const float*)d_in[0];
    const float* wg = (const float*)d_in[1];
    const float* w1 = (const float*)d_in[2];
    const float* b1 = (const float*)d_in[3];
    const float* w2 = (const float*)d_in[4];
    const float* b2 = (const float*)d_in[5];
    float* out = (float*)d_out;

    char* ws = (char*)d_ws;
    size_t off = 0;
    int*   gidx       = (int*)(ws + off);   off += (size_t)NTOK * TOPK * 4;
    float* gval       = (float*)(ws + off); off += (size_t)NTOK * TOPK * 4;
    int*   pos        = (int*)(ws + off);   off += (size_t)NTOK * TOPK * 4;
    int*   slot_token = (int*)(ws + off);   off += (size_t)NEXP * CAP * 4;
    unsigned short* xbf = (unsigned short*)(ws + off);
    off += (size_t)NTOK * DMODEL * 2;                    // 16.8 MB
    unsigned short* w1t = (unsigned short*)(ws + off);
    off += (size_t)NEXP * DMODEL * HFF * 2;              // 67 MB
    unsigned short* w2t = (unsigned short*)(ws + off);
    off += (size_t)NEXP * HFF * DMODEL * 2;              // 67 MB
    unsigned short* h = (unsigned short*)(ws + off);
    off += (size_t)NEXP * CAP * HFF * 2;                 // 134 MB
    float* y = (float*)(ws + off);                       // 67 MB

    cvt_x_kernel<<<(NTOK * DMODEL) / 1024, 256, 0, stream>>>(x, xbf);
    tconv_kernel<<<dim3(HFF / 32, DMODEL / 32, NEXP), 256, 0, stream>>>(
        w1, w1t, DMODEL, HFF);
    tconv_kernel<<<dim3(DMODEL / 32, HFF / 32, NEXP), 256, 0, stream>>>(
        w2, w2t, HFF, DMODEL);
    gate_kernel<<<NTOK / 8, 256, 0, stream>>>(x, wg, gidx, gval);
    scan_kernel<<<1, 256, 0, stream>>>(gidx, gval, pos, slot_token);
    gemm1_kernel<<<dim3(HFF / NT, CAP / MT, NEXP), 256, 0, stream>>>(
        xbf, w1t, b1, slot_token, h);
    gemm2_kernel<<<dim3(DMODEL / NT, CAP / MT, NEXP), 256, 0, stream>>>(
        h, w2t, b2, y);
    combine_kernel<<<NTOK, 256, 0, stream>>>(y, gidx, gval, pos, out);
}
